// SwinBlock_884763263599
// MI455X (gfx1250) — matrix operand order repaired and checked by reading, not compiled, not run on hardware
//
#include <hip/hip_runtime.h>
#include <math.h>

// ---------------------------------------------------------------------------
// Types for CDNA5 WMMA (wave32, bf16 in / fp32 accumulate)
// ---------------------------------------------------------------------------
typedef __bf16 bf16;
typedef __bf16 bf16x16 __attribute__((ext_vector_type(16)));
typedef float  f32x4   __attribute__((ext_vector_type(4)));
typedef float  v8f     __attribute__((ext_vector_type(8)));

union FragU { bf16x16 v; f32x4 q[2]; };

__device__ __forceinline__ v8f vzero() {
  v8f z;
#pragma unroll
  for (int i = 0; i < 8; ++i) z[i] = 0.f;
  return z;
}

__device__ __forceinline__ v8f wmma_bf16(bf16x16 a, bf16x16 b, v8f c) {
  // v_wmma_f32_16x16x32_bf16  D = A(16x32) * B(32x16) + C
  return __builtin_amdgcn_wmma_f32_16x16x32_bf16(false, a, false, b,
                                                 (short)0, c, false, false);
}

// A fragment (16x32 bf16, row-major source, stride lda elements).
// Layout: lanes 0-15 -> M=lane, K 0..15; lanes 16-31 -> K 16..31 (contiguous, matches B)
__device__ __forceinline__ bf16x16 frag_A(const bf16* base, int lda, int lane) {
  int m = lane & 15, half = (lane >> 4) & 1;
  const bf16* p = base + (size_t)m * lda + half * 16;
  FragU f;
  f.q[0] = *(const f32x4*)p;        // K = half*16 + 0..7
  f.q[1] = *(const f32x4*)(p + 8);  // K = half*16 + 8..15
  return f.v;
}

// B fragment representing X^T where X is row-major [N][K] (stride ldx).
// B layout used here (same K order as frag_A): lane -> N = lane&15, K = (lane>>4)*16 + i  (16 contiguous K per lane)
__device__ __forceinline__ bf16x16 frag_BT(const bf16* base, int ldx, int lane) {
  int n = lane & 15, half = (lane >> 4) & 1;
  const bf16* p = base + (size_t)n * ldx + half * 16;
  FragU f;
  f.q[0] = *(const f32x4*)p;
  f.q[1] = *(const f32x4*)(p + 8);
  return f.v;
}

// B fragment from pre-packed weight tile: 32 lanes x 16 halfs contiguous.
__device__ __forceinline__ bf16x16 frag_Bpacked(const bf16* tile, int lane) {
  const bf16* p = tile + lane * 16;
  FragU f;
  f.q[0] = *(const f32x4*)p;
  f.q[1] = *(const f32x4*)(p + 8);
  return f.v;
}

// ---------------------------------------------------------------------------
// CDNA5 async global->LDS copy (ASYNCcnt path, CDNA5 ISA)
// dst_lds: wave-relative LDS byte offset (low 32 bits of flat LDS pointer)
// ---------------------------------------------------------------------------
__device__ __forceinline__ void async_copy_b128(unsigned dst_lds,
                                                const void* src) {
  asm volatile("global_load_async_to_lds_b128 %0, %1, off"
               :: "v"(dst_lds), "v"(src)
               : "memory");
}
__device__ __forceinline__ void async_wait0() {
  asm volatile("s_wait_asynccnt 0x0" ::: "memory");
}
__device__ __forceinline__ unsigned lds_off(const void* p) {
  return (unsigned)(size_t)p;   // low 32 bits of flat LDS addr = LDS offset
}

// ---------------------------------------------------------------------------
// Reductions (wave32)
// ---------------------------------------------------------------------------
__device__ __forceinline__ float red_sum32(float v) {
#pragma unroll
  for (int m = 1; m <= 16; m <<= 1) v += __shfl_xor(v, m, 32);
  return v;
}
__device__ __forceinline__ float red_sum16(float v) {
#pragma unroll
  for (int m = 1; m <= 8; m <<= 1) v += __shfl_xor(v, m, 32);
  return v;
}
__device__ __forceinline__ float red_max16(float v) {
#pragma unroll
  for (int m = 1; m <= 8; m <<= 1) v = fmaxf(v, __shfl_xor(v, m, 32));
  return v;
}

// ---------------------------------------------------------------------------
// Problem constants
// ---------------------------------------------------------------------------
#define CB   32       // batch
#define CHW  56
#define CC   128
#define CWS  7
#define CSS  3
#define CNH  4
#define CHD  32
#define NPAD 64       // padded tokens per window (49 -> 64)
#define BW   (CB * 64)               // 2048 windows
#define SCALE 0.17677669529663687f   // 32^-0.5

// ---------------------------------------------------------------------------
// Weight packing: fp32 [K][N] row-major -> bf16 WMMA B tiles
// tile (ntile,kt): element (lane,i) = W[kt*32 + (lane>>4)*16 + i][ntile*16 + (lane&15)]
// ---------------------------------------------------------------------------
__global__ void k_pack(const float* __restrict__ W, bf16* __restrict__ out,
                       int K, int N) {
  int idx = blockIdx.x * 256 + threadIdx.x;
  int Kt = K >> 5;
  int total = Kt * (N >> 4) * 512;
  if (idx >= total) return;
  int li = idx & 511;
  int tile = idx >> 9;
  int kt = tile % Kt, ntile = tile / Kt;
  int lane = li >> 4, i = li & 15;
  int k = kt * 32 + ((lane >> 4) << 4) + i;
  int n = (ntile << 4) + (lane & 15);
  out[idx] = (bf16)W[(size_t)k * N + n];
}

// ---------------------------------------------------------------------------
// LN1 + cyclic shift(-3,-3) + window partition -> bf16 [2048][64][128] (pad=0)
// ---------------------------------------------------------------------------
__global__ void k_ln1_shift(const float* __restrict__ x,
                            const float* __restrict__ w,
                            const float* __restrict__ b,
                            bf16* __restrict__ yw) {
  int lane = threadIdx.x & 31, wv = threadIdx.x >> 5;
  int task = blockIdx.x * 8 + wv;          // win*64 + tok
  int win = task >> 6, tok = task & 63;
  bf16* dst = yw + (size_t)task * CC + lane * 4;
  if (tok >= 49) {
    bf16 z = (bf16)0.0f;
    dst[0] = z; dst[1] = z; dst[2] = z; dst[3] = z;
    return;
  }
  int bi = win >> 6, wy = (win >> 3) & 7, wx = win & 7;
  int hh = (wy * CWS + tok / CWS + CSS) % CHW;
  int ww = (wx * CWS + tok % CWS + CSS) % CHW;
  const float* row = x + ((size_t)bi * (CHW * CHW) + hh * CHW + ww) * CC;
  f32x4 v = *(const f32x4*)(row + lane * 4);
  float s = red_sum32(v[0] + v[1] + v[2] + v[3]);
  float mu = s * (1.0f / CC);
  f32x4 d;
#pragma unroll
  for (int i = 0; i < 4; ++i) d[i] = v[i] - mu;
  float q = red_sum32(d[0]*d[0] + d[1]*d[1] + d[2]*d[2] + d[3]*d[3]);
  float rstd = rsqrtf(q * (1.0f / CC) + 1e-5f);
#pragma unroll
  for (int i = 0; i < 4; ++i) {
    int c = lane * 4 + i;
    dst[i] = (bf16)(d[i] * rstd * w[c] + b[c]);
  }
}

// ---------------------------------------------------------------------------
// LN2 over d_out rows -> bf16 [100352][128]
// ---------------------------------------------------------------------------
__global__ void k_ln2(const float* __restrict__ xr,
                      const float* __restrict__ w,
                      const float* __restrict__ b,
                      bf16* __restrict__ out) {
  int lane = threadIdx.x & 31, wv = threadIdx.x >> 5;
  size_t row = (size_t)blockIdx.x * 8 + wv;
  const float* p = xr + row * CC;
  f32x4 v = *(const f32x4*)(p + lane * 4);
  float s = red_sum32(v[0] + v[1] + v[2] + v[3]);
  float mu = s * (1.0f / CC);
  f32x4 d;
#pragma unroll
  for (int i = 0; i < 4; ++i) d[i] = v[i] - mu;
  float q = red_sum32(d[0]*d[0] + d[1]*d[1] + d[2]*d[2] + d[3]*d[3]);
  float rstd = rsqrtf(q * (1.0f / CC) + 1e-5f);
  bf16* dst = out + row * CC + lane * 4;
#pragma unroll
  for (int i = 0; i < 4; ++i) {
    int c = lane * 4 + i;
    dst[i] = (bf16)(d[i] * rstd * w[c] + b[c]);
  }
}

// ---------------------------------------------------------------------------
// Generic WMMA GEMM: block = 8 waves, each wave -> 16(M) x 64(N) output strip.
// B strip (4 ntiles x Kt, contiguous in packed layout) is async-staged to LDS
// once per block, then all waves read fragments via ds_load_b128.
// MODE 0: qkv epilogue  -> q (scaled), k, vT          (o0,o1,o2)
// MODE 1: proj epilogue -> window-reverse + unshift + residual into fout
// MODE 2: fc1 epilogue  -> exact GELU, bf16 out       (o0, ld 512)
// MODE 3: fc2 epilogue  -> fout[M*128+N] += acc + bias
// ---------------------------------------------------------------------------
template <int MODE>
__global__ __launch_bounds__(256)
void k_gemm(const bf16* __restrict__ A, int lda, int Kt,
            const bf16* __restrict__ Bp, const float* __restrict__ bias,
            float* __restrict__ fout, const float* __restrict__ res,
            bf16* __restrict__ o0, bf16* __restrict__ o1,
            bf16* __restrict__ o2) {
  extern __shared__ __align__(16) bf16 Bsh[];   // 4*Kt KB
  int lane = threadIdx.x & 31, wv = threadIdx.x >> 5;
  int half = lane >> 4, l16 = lane & 15;
  size_t mbase = ((size_t)blockIdx.x * 8 + wv) * 16;
  int nbase = blockIdx.y * 64;

  // ---- async-stage packed B strip into LDS (each byte copied once) ----
  {
    const char* gsrc = (const char*)(Bp + (((size_t)blockIdx.y * 4 * Kt) << 9));
    unsigned lbase = lds_off(Bsh);
    for (int j = 0; j < Kt; ++j) {
      unsigned off = (unsigned)((j * 256 + threadIdx.x) * 16);
      async_copy_b128(lbase + off, gsrc + off);
    }
    async_wait0();
    __syncthreads();
  }

  v8f acc[4] = {vzero(), vzero(), vzero(), vzero()};
  const bf16* Ab = A + mbase * lda;

  for (int kt = 0; kt < Kt; ++kt) {
    bf16x16 a = frag_A(Ab + kt * 32, lda, lane);
#pragma unroll
    for (int nt = 0; nt < 4; ++nt) {
      const bf16* tile = Bsh + (((size_t)(nt * Kt + kt)) << 9);
      bf16x16 b = frag_Bpacked(tile, lane);
      acc[nt] = wmma_bf16(a, b, acc[nt]);
    }
  }

#pragma unroll
  for (int nt = 0; nt < 4; ++nt) {
    int N = nbase + nt * 16 + l16;
    float bn = bias[N];
#pragma unroll
    for (int r = 0; r < 8; ++r) {
      size_t M = mbase + r + half * 8;   // C layout: VGPR r -> M=r (lo half), r+8 (hi half)
      float v = acc[nt][r] + bn;
      if (MODE == 0) {
        int s = N >> 7, h = (N >> 5) & 3, d = N & 31;
        size_t win = M >> 6, tok = M & 63;
        if (s == 0) {
          v *= SCALE;
          o0[((win * CNH + h) * NPAD + tok) * CHD + d] = (bf16)v;   // q
        } else if (s == 1) {
          o1[((win * CNH + h) * NPAD + tok) * CHD + d] = (bf16)v;   // k
        } else {
          o2[((win * CNH + h) * CHD + d) * NPAD + tok] = (bf16)v;   // v^T
        }
      } else if (MODE == 1) {
        size_t win = M >> 6;
        int tok = (int)(M & 63);
        if (tok < 49) {
          size_t bi = win >> 6;
          int wy = (int)((win >> 3) & 7), wx = (int)(win & 7);
          int hh = (wy * CWS + tok / CWS + CSS) % CHW;
          int ww = (wx * CWS + tok % CWS + CSS) % CHW;
          size_t idx = (bi * (CHW * CHW) + (size_t)hh * CHW + ww) * CC + N;
          fout[idx] = res[idx] + v;
        }
      } else if (MODE == 2) {
        float g = 0.5f * v * (1.0f + erff(v * 0.70710678118654752f));
        o0[M * 512 + N] = (bf16)g;
      } else {
        fout[M * CC + N] += v;
      }
    }
  }
}

// ---------------------------------------------------------------------------
// Shifted-window attention: one block per window (2048 blocks), 8 waves.
// K and V^T panels for the window (16 KB each) async-staged to LDS.
// 16 tasks (head, 16-row tile). QK^T (4 wmma, K=32), bias+mask+softmax,
// P->LDS bf16, P@V (4 wmma, K=64), store bf16 [win][tok][h*32+d].
// ---------------------------------------------------------------------------
__global__ __launch_bounds__(256)
void k_attn(const bf16* __restrict__ q, const bf16* __restrict__ k,
            const bf16* __restrict__ vT, const float* __restrict__ ptab,
            bf16* __restrict__ attn_out) {
  __shared__ __align__(16) bf16 P[8][16 * 64];   // 16 KB per-wave P tiles
  __shared__ __align__(16) bf16 Ksh[CNH * NPAD * CHD];  // 16 KB
  __shared__ __align__(16) bf16 Vsh[CNH * CHD * NPAD];  // 16 KB
  int lane = threadIdx.x & 31, wv = threadIdx.x >> 5;
  int half = lane >> 4, l16 = lane & 15;
  int win = blockIdx.x;
  int wy = (win >> 3) & 7, wx = win & 7;

  // ---- async-stage K and V^T panels (4 heads, contiguous) -------------
  {
    const char* gk = (const char*)(k  + (size_t)win * CNH * NPAD * CHD);
    const char* gv = (const char*)(vT + (size_t)win * CNH * CHD * NPAD);
    unsigned lk = lds_off(Ksh), lv = lds_off(Vsh);
#pragma unroll
    for (int j = 0; j < 4; ++j) {           // 4 * 256 * 16B = 16 KB each
      unsigned off = (unsigned)((j * 256 + threadIdx.x) * 16);
      async_copy_b128(lk + off, gk + off);
      async_copy_b128(lv + off, gv + off);
    }
    async_wait0();
    __syncthreads();
  }

  for (int t = wv; t < 16; t += 8) {
    int h = t >> 2, mt = t & 3;
    const bf16* qh = q + ((size_t)win * CNH + h) * (NPAD * CHD);
    const bf16* kh = Ksh + h * (NPAD * CHD);
    const bf16* vh = Vsh + h * (CHD * NPAD);

    // ---- S = q @ k^T  rows [mt*16, +16) x cols [0,64) -----------------
    bf16x16 a = frag_A(qh + (size_t)(mt * 16) * CHD, CHD, lane);
    v8f acc[4] = {vzero(), vzero(), vzero(), vzero()};
#pragma unroll
    for (int nt = 0; nt < 4; ++nt) {
      bf16x16 bb = frag_BT(kh + (size_t)(nt * 16) * CHD, CHD, lane);
      acc[nt] = wmma_bf16(a, bb, acc[nt]);
    }

    // ---- bias + shift mask + padding mask, then softmax over cols -----
    float rowmax[8], rowsum[8];
#pragma unroll
    for (int r = 0; r < 8; ++r) {
      int mrow = mt * 16 + r + half * 8;     // query token
      bool mvalid = mrow < 49;
      int my = 0, mx = 0, rm = 0;
      if (mvalid) {
        my = mrow / 7; mx = mrow % 7;
        int py = wy * CWS + my, px = wx * CWS + mx;
        rm = (py < 49 ? 0 : (py < 53 ? 1 : 2)) * 3 +
             (px < 49 ? 0 : (px < 53 ? 1 : 2));
      }
      float m4 = -1e30f;
#pragma unroll
      for (int nt = 0; nt < 4; ++nt) {
        int nc = nt * 16 + l16;              // key token
        float v = acc[nt][r];
        if (nc >= 49) {
          v = -1e30f;                         // padded key -> masked out
        } else if (mvalid) {
          int ny = nc / 7, nx = nc % 7;
          v += ptab[((my - ny + 6) * 13 + (mx - nx + 6)) * CNH + h];
          int py = wy * CWS + ny, px = wx * CWS + nx;
          int rn = (py < 49 ? 0 : (py < 53 ? 1 : 2)) * 3 +
                   (px < 49 ? 0 : (px < 53 ? 1 : 2));
          if (rn != rm) v += -100.0f;
        } else {
          v = 0.0f;                           // padded query row: keep finite
        }
        acc[nt][r] = v;
        m4 = fmaxf(m4, v);
      }
      rowmax[r] = red_max16(m4);
    }
#pragma unroll
    for (int r = 0; r < 8; ++r) {
      float s4 = 0.f;
#pragma unroll
      for (int nt = 0; nt < 4; ++nt) {
        float e = __expf(acc[nt][r] - rowmax[r]);
        acc[nt][r] = e;
        s4 += e;
      }
      rowsum[r] = red_sum16(s4);
    }
#pragma unroll
    for (int r = 0; r < 8; ++r) {
      float inv = 1.0f / rowsum[r];
      int lm = r + half * 8;
#pragma unroll
      for (int nt = 0; nt < 4; ++nt)
        P[wv][lm * 64 + nt * 16 + l16] = (bf16)(acc[nt][r] * inv);
    }
    __syncthreads();

    // ---- O = P @ V : [16 x 32], K=64 (2 k-steps) -----------------------
    v8f o[2] = {vzero(), vzero()};
#pragma unroll
    for (int s = 0; s < 2; ++s) {
      bf16x16 pa = frag_A(&P[wv][s * 32], 64, lane);
#pragma unroll
      for (int n2 = 0; n2 < 2; ++n2) {
        bf16x16 vb = frag_BT(vh + (size_t)(n2 * 16) * NPAD + s * 32, NPAD, lane);
        o[n2] = wmma_bf16(pa, vb, o[n2]);
      }
    }
#pragma unroll
    for (int n2 = 0; n2 < 2; ++n2)
#pragma unroll
      for (int r = 0; r < 8; ++r) {
        int tokr = mt * 16 + r + half * 8;
        attn_out[((size_t)win * NPAD + tokr) * CC + h * CHD + n2 * 16 + l16] =
            (bf16)(o[n2][r]);
      }
    __syncthreads();
  }
}

// ---------------------------------------------------------------------------
// Host-side orchestration
// ---------------------------------------------------------------------------
extern "C" void kernel_launch(void* const* d_in, const int* in_sizes, int n_in,
                              void* d_out, int out_size, void* d_ws,
                              size_t ws_size, hipStream_t stream) {
  (void)in_sizes; (void)n_in; (void)out_size; (void)ws_size;
  const float* x     = (const float*)d_in[0];
  const float* n1w   = (const float*)d_in[1];
  const float* n1b   = (const float*)d_in[2];
  const float* qkvw  = (const float*)d_in[3];
  const float* qkvb  = (const float*)d_in[4];
  const float* projw = (const float*)d_in[5];
  const float* projb = (const float*)d_in[6];
  const float* ptab  = (const float*)d_in[7];
  const float* w1    = (const float*)d_in[8];
  const float* b1    = (const float*)d_in[9];
  const float* w2    = (const float*)d_in[10];
  const float* b2    = (const float*)d_in[11];
  const float* n2w   = (const float*)d_in[12];
  const float* n2b   = (const float*)d_in[13];
  float* out = (float*)d_out;
  char* ws = (char*)d_ws;

  // workspace layout (bytes); total ~194 MB, h1 aliases dead attention bufs
  const size_t SZ = 33554432ull;               // 2048*64*128*2B
  bf16* yw   = (bf16*)(ws + 0);
  bf16* q    = (bf16*)(ws + SZ);
  bf16* kk   = (bf16*)(ws + 2 * SZ);
  bf16* vT   = (bf16*)(ws + 3 * SZ);
  bf16* attn = (bf16*)(ws + 4 * SZ);
  bf16* xb   = (bf16*)(ws + 5 * SZ);           // 25.7 MB
  bf16* h1   = (bf16*)(ws + 0);                // 102.8 MB, aliases yw/q/kk (dead)
  bf16* pqkv = (bf16*)(ws + 5 * SZ + 25690112ull);
  bf16* pprj = pqkv + 128 * 384;
  bf16* pw1  = pprj + 128 * 128;
  bf16* pw2  = pw1 + 128 * 512;

  // weight repack (tiny, amortized)
  k_pack<<<192, 256, 0, stream>>>(qkvw, pqkv, 128, 384);
  k_pack<<< 64, 256, 0, stream>>>(projw, pprj, 128, 128);
  k_pack<<<256, 256, 0, stream>>>(w1, pw1, 128, 512);
  k_pack<<<256, 256, 0, stream>>>(w2, pw2, 512, 128);

  // 1) LN1 + shift + window partition (pad 49->64)
  k_ln1_shift<<<16384, 256, 0, stream>>>(x, n1w, n1b, yw);

  // 2) QKV GEMM: [131072 x 128] @ [128 x 384]   (16 KB LDS B-stage)
  k_gemm<0><<<dim3(1024, 6), 256, 16384, stream>>>(yw, 128, 4, pqkv, qkvb,
                                                   nullptr, nullptr, q, kk, vT);

  // 3) windowed attention
  k_attn<<<2048, 256, 0, stream>>>(q, kk, vT, ptab, attn);

  // 4) proj GEMM + window reverse + unshift + residual -> d_out = x'
  k_gemm<1><<<dim3(1024, 2), 256, 16384, stream>>>(attn, 128, 4, pprj, projb,
                                                   out, x, nullptr, nullptr, nullptr);

  // 5) LN2 -> bf16
  k_ln2<<<12544, 256, 0, stream>>>(out, n2w, n2b, xb);

  // 6) FC1 + GELU: [100352 x 128] @ [128 x 512]
  k_gemm<2><<<dim3(784, 8), 256, 16384, stream>>>(xb, 128, 4, pw1, b1,
                                                  nullptr, nullptr, h1, nullptr, nullptr);

  // 7) FC2 accumulate: d_out += [100352 x 512] @ [512 x 128] + b2  (64 KB LDS)
  k_gemm<3><<<dim3(784, 2), 256, 65536, stream>>>(h1, 512, 16, pw2, b2,
                                                  out, nullptr, nullptr, nullptr, nullptr);
}
